// FEM_32409823216129
// MI455X (gfx1250) — matrix operand order repaired and checked by reading, not compiled, not run on hardware
//
#include <hip/hip_runtime.h>
#include <hip/hip_bf16.h>

typedef __bf16 bf16;
typedef __attribute__((ext_vector_type(16))) __bf16 v16bf;
typedef __attribute__((ext_vector_type(8)))  __bf16 v8bf;
typedef __attribute__((ext_vector_type(8)))  float  v8f;

#define Bb 16
#define Cc 320
#define Nn 4096
#define ICic 128
#define NWG_K3 ((Nn / 64) * Bb)   // 1024 workgroups in apply_trans
#define KCHUNK 8                  // split-K chunks in attention logits

// ---------------------------------------------------------------------------
// WMMA fragment loaders (CDNA5 wave32, V_WMMA_F32_16X16X32_BF16)
// A (16x32, MxK) from row-major [M][K]: lane row = lane&15,
//   halves 0..7  -> K = 8*(lane>>4) + 0..7      (16B contiguous)
//   halves 8..15 -> K = 16 + 8*(lane>>4) + 0..7 (16B contiguous)
// B (32x16, KxN) from row-major [N][K] (K-contiguous per column):
//   lane col = lane&15, same K mapping as A (halves 0..7 -> 8*(lane>>4)+0..7,
//   halves 8..15 -> 16 + 8*(lane>>4) + 0..7)
// ---------------------------------------------------------------------------
__device__ __forceinline__ v16bf frag_a(const bf16* base, int ld, int lane) {
  const bf16* p = base + (lane & 15) * ld + ((lane >> 4) << 3);
  v8bf lo = *(const v8bf*)p;
  v8bf hi = *(const v8bf*)(p + 16);
  v16bf r;
#pragma unroll
  for (int i = 0; i < 8; ++i) { r[i] = lo[i]; r[i + 8] = hi[i]; }
  return r;
}

__device__ __forceinline__ v16bf frag_b(const bf16* base, int ld, int lane) {
  const bf16* p = base + (lane & 15) * ld + ((lane >> 4) << 3);
  v8bf lo = *(const v8bf*)p;
  v8bf hi = *(const v8bf*)(p + 16);
  v16bf r;
#pragma unroll
  for (int i = 0; i < 8; ++i) { r[i] = lo[i]; r[i + 8] = hi[i]; }
  return r;
}

__device__ __forceinline__ v8f wmma_bf16(v16bf a, v16bf b, v8f c) {
  return __builtin_amdgcn_wmma_f32_16x16x32_bf16(false, a, false, b, (short)0, c,
                                                 false, false);
}

// ---------------------------------------------------------------------------
// K0: transpose fp32 [rows][cols] -> bf16 [cols][rows]
// ---------------------------------------------------------------------------
__global__ void transpose_bf16_kernel(const float* __restrict__ src,
                                      bf16* __restrict__ dst, int rows, int cols) {
  int i = blockIdx.x * 256 + threadIdx.x;
  if (i >= rows * cols) return;
  int r = i / cols, c = i % cols;
  dst[c * rows + r] = (bf16)src[i];
}

// ---------------------------------------------------------------------------
// K1: two channel projections of one input X [B][320][4096].
//   out0T[b][n][128] (token-major)   out1[b][128][n] (channel-major)
// Grid (N/64, B), 256 threads. Wave w owns (proj, mtile=w); K-outer loop with
// 4 token-tile accumulators; A fragment double-buffered (next K-step load in
// flight while current WMMAs execute).
// ---------------------------------------------------------------------------
__global__ __launch_bounds__(256) void proj_kernel(
    const float* __restrict__ X,
    const bf16* __restrict__ W0T, const float* __restrict__ bias0,
    bf16* __restrict__ out0T,
    const bf16* __restrict__ W1T, const float* __restrict__ bias1,
    bf16* __restrict__ out1) {
  __shared__ bf16 xl[64 * 328];  // [token][channel], stride 328 (bank-spread)
  const int b = blockIdx.y;
  const int n0 = blockIdx.x * 64;
  const int tid = threadIdx.x;
  const float* Xb = X + (size_t)b * Cc * Nn;
  {
    int tn = tid & 63;
    int c0 = tid >> 6;
    for (int c = c0; c < Cc; c += 4)
      xl[tn * 328 + c] = (bf16)Xb[(size_t)c * Nn + n0 + tn];
  }
  __syncthreads();
  const int wave = __builtin_amdgcn_readfirstlane(tid >> 5);  // scalar
  const int lane = tid & 31;
  const int g = lane >> 4, l15 = lane & 15;
  const int mt = wave;  // mtile 0..7
  for (int proj = 0; proj < 2; ++proj) {
    const bf16* WT = proj ? W1T : W0T;
    const bf16* arow = WT + (mt * 16) * Cc;
    v8f acc[4] = {};
    v16bf a = frag_a(arow, Cc, lane);
#pragma unroll
    for (int k0 = 0; k0 < Cc - 32; k0 += 32) {
      v16bf an = frag_a(arow + k0 + 32, Cc, lane);  // prefetch next K step
#pragma unroll
      for (int nt = 0; nt < 4; ++nt)
        acc[nt] = wmma_bf16(a, frag_b(xl + (nt * 16) * 328 + k0, 328, lane),
                            acc[nt]);
      a = an;
    }
#pragma unroll
    for (int nt = 0; nt < 4; ++nt)
      acc[nt] = wmma_bf16(a, frag_b(xl + (nt * 16) * 328 + (Cc - 32), 328, lane),
                          acc[nt]);
    int i0 = mt * 16 + g * 8;
    if (proj == 0) {
#pragma unroll
      for (int nt = 0; nt < 4; ++nt) {
        int n = n0 + nt * 16 + l15;
        v8bf pk;
#pragma unroll
        for (int r = 0; r < 8; ++r) pk[r] = (bf16)(acc[nt][r] + bias0[i0 + r]);
        *(v8bf*)(out0T + ((size_t)b * Nn + n) * ICic + i0) = pk;
      }
    } else {
#pragma unroll
      for (int nt = 0; nt < 4; ++nt) {
        int n = n0 + nt * 16 + l15;
#pragma unroll
        for (int r = 0; r < 8; ++r)
          out1[((size_t)b * ICic + i0 + r) * Nn + n] =
              (bf16)(acc[nt][r] + bias1[i0 + r]);
      }
    }
  }
}

// ---------------------------------------------------------------------------
// K2a: split-K attention logits. Grid (B, KCHUNK); chunk = 512 tokens.
// Wave w owns a 32x64 block (2 row-tiles x 4 col-tiles): per K step
// 4 A-loads + 8 B-loads feed 8 WMMAs (2x fragment reuse), A fragments
// double-buffered. Partials -> A_part[b][chunk] (deterministic).
// ---------------------------------------------------------------------------
__global__ __launch_bounds__(256) void attn_logits_kernel(
    const bf16* __restrict__ kx, const bf16* __restrict__ qx,
    float* __restrict__ A_part) {
  int b = blockIdx.x, ch = blockIdx.y, tid = threadIdx.x;
  int wave = __builtin_amdgcn_readfirstlane(tid >> 5);
  int lane = tid & 31, g = lane >> 4, l15 = lane & 15;
  int rp = wave >> 1;   // row pair 0..3 -> rows rp*32 and rp*32+16
  int chh = wave & 1;   // column half: cols chh*64 .. chh*64+63
  const bf16* kb = kx + (size_t)b * ICic * Nn;
  const bf16* qb = qx + (size_t)b * ICic * Nn + (size_t)(chh * 64) * Nn;
  float* Ap = A_part + ((size_t)b * KCHUNK + ch) * ICic * ICic;
  const bf16* ar0 = kb + (size_t)(rp * 32) * Nn;
  const bf16* ar1 = kb + (size_t)(rp * 32 + 16) * Nn;
  const int kbeg = ch * (Nn / KCHUNK), kend = kbeg + (Nn / KCHUNK);
  v8f acc[2][4] = {};
  v16bf a0 = frag_a(ar0 + kbeg, Nn, lane);
  v16bf a1 = frag_a(ar1 + kbeg, Nn, lane);
  for (int k0 = kbeg; k0 < kend - 32; k0 += 32) {
    v16bf an0 = frag_a(ar0 + k0 + 32, Nn, lane);  // prefetch next K step
    v16bf an1 = frag_a(ar1 + k0 + 32, Nn, lane);
#pragma unroll
    for (int jt = 0; jt < 4; ++jt) {
      v16bf bb = frag_b(qb + (size_t)(jt * 16) * Nn + k0, Nn, lane);
      acc[0][jt] = wmma_bf16(a0, bb, acc[0][jt]);
      acc[1][jt] = wmma_bf16(a1, bb, acc[1][jt]);
    }
    a0 = an0;
    a1 = an1;
  }
#pragma unroll
  for (int jt = 0; jt < 4; ++jt) {
    v16bf bb = frag_b(qb + (size_t)(jt * 16) * Nn + (kend - 32), Nn, lane);
    acc[0][jt] = wmma_bf16(a0, bb, acc[0][jt]);
    acc[1][jt] = wmma_bf16(a1, bb, acc[1][jt]);
  }
#pragma unroll
  for (int rr = 0; rr < 2; ++rr)
#pragma unroll
    for (int jt = 0; jt < 4; ++jt)
#pragma unroll
      for (int r = 0; r < 8; ++r)
        Ap[(rp * 32 + rr * 16 + r + 8 * g) * ICic + chh * 64 + jt * 16 + l15] =
            acc[rr][jt][r];
}

// K2b: sum split-K partials -> A[b][128][128]
__global__ void reduce_A_kernel(const float* __restrict__ Ap,
                                float* __restrict__ A) {
  int i = blockIdx.x * 256 + threadIdx.x;
  if (i >= Bb * ICic * ICic) return;
  int b = i >> 14, rc = i & 16383;
  const float* p = Ap + (size_t)b * KCHUNK * 16384 + rc;
  float sum = 0.f;
#pragma unroll
  for (int ch = 0; ch < KCHUNK; ++ch) sum += p[ch * 16384];
  A[i] = sum;
}

// ---------------------------------------------------------------------------
// K2c: att_s = softmax(A, rows), att_q = softmax(A^T, rows). Grid (B, 2).
// ---------------------------------------------------------------------------
__global__ __launch_bounds__(128) void softmax_kernel(
    const float* __restrict__ A, bf16* __restrict__ att_s,
    bf16* __restrict__ att_q) {
  int b = blockIdx.x, t = threadIdx.x;
  bool colmode = blockIdx.y != 0;
  const float* Ab = A + (size_t)b * ICic * ICic;
  int stride = colmode ? ICic : 1;
  const float* row = colmode ? (Ab + t) : (Ab + t * ICic);
  float mx = -1e30f;
  for (int i = 0; i < ICic; ++i) mx = fmaxf(mx, row[i * stride]);
  float sm = 0.f;
  for (int i = 0; i < ICic; ++i) sm += __expf(row[i * stride] - mx);
  float inv = 1.f / sm;
  bf16* out = (colmode ? att_q : att_s) + (size_t)b * ICic * ICic + t * ICic;
  for (int i = 0; i < ICic; ++i)
    out[i] = (bf16)(__expf(row[i * stride] - mx) * inv);
}

// ---------------------------------------------------------------------------
// K3: fused p = att @ v (128xN, K=128) ->(LDS)-> T = W^T @ p + bias (320xN),
// plus deterministic per-WG BN sum/sumsq partials (single writer per slot,
// no float atomics). Grid (N/64, B), 256 threads.
// ---------------------------------------------------------------------------
__global__ __launch_bounds__(256) void apply_trans_kernel(
    const bf16* __restrict__ att_s, const bf16* __restrict__ att_q,
    const bf16* __restrict__ vsT, const bf16* __restrict__ vqT,
    const bf16* __restrict__ WtsT, const float* __restrict__ bts,
    const bf16* __restrict__ WtqT, const float* __restrict__ btq,
    bf16* __restrict__ T_s, bf16* __restrict__ T_q,
    float* __restrict__ stats_part /* [1280][NWG_K3] */) {
  __shared__ bf16 pl[64 * 136];   // p_s tile, token-major [tok][128], pad->136
  __shared__ bf16 ql[64 * 136];   // q_s tile
  __shared__ float st[4 * Cc];    // sum_s,sq_s,sum_q,sq_q (single writer each)
  int b = blockIdx.y, n0 = blockIdx.x * 64, tid = threadIdx.x;
  int wave = __builtin_amdgcn_readfirstlane(tid >> 5);
  int lane = tid & 31, g = lane >> 4, l15 = lane & 15;
  // ---- phase 1: attention apply (out=cc, mtile=wave, 4 nt accumulators)
  for (int cc = 0; cc < 2; ++cc) {
    int mt = wave;
    const bf16* att = (cc ? att_q : att_s) + (size_t)b * ICic * ICic +
                      (mt * 16) * ICic;
    const bf16* v = (cc ? vqT : vsT) + ((size_t)b * Nn + n0) * ICic;
    v8f acc[4] = {};
    v16bf a = frag_a(att, ICic, lane);
#pragma unroll
    for (int k0 = 0; k0 < ICic - 32; k0 += 32) {
      v16bf an = frag_a(att + k0 + 32, ICic, lane);
#pragma unroll
      for (int nt = 0; nt < 4; ++nt)
        acc[nt] = wmma_bf16(a, frag_b(v + (nt * 16) * ICic + k0, ICic, lane),
                            acc[nt]);
      a = an;
    }
#pragma unroll
    for (int nt = 0; nt < 4; ++nt)
      acc[nt] = wmma_bf16(
          a, frag_b(v + (nt * 16) * ICic + (ICic - 32), ICic, lane), acc[nt]);
    bf16* dst = cc ? ql : pl;
    int i0 = mt * 16 + 8 * g;
#pragma unroll
    for (int nt = 0; nt < 4; ++nt) {
      int tok = nt * 16 + l15;
      v8bf pk;
#pragma unroll
      for (int r = 0; r < 8; ++r) pk[r] = (bf16)acc[nt][r];
      *(v8bf*)(dst + tok * 136 + i0) = pk;
    }
  }
  __syncthreads();
  // ---- phase 2: Trans linear + BN partials. 40 (out,mt) combos, 5 per wave.
  for (int ccj = 0; ccj < 5; ++ccj) {
    int combo = wave + (ccj << 3);       // 0..39
    int outq = combo >= 20;
    int mt = outq ? combo - 20 : combo;  // 0..19
    const bf16* WT = (outq ? WtqT : WtsT) + (mt * 16) * ICic;
    const bf16* pb = outq ? ql : pl;
    v8f acc[4] = {};
    v16bf a = frag_a(WT, ICic, lane);
#pragma unroll
    for (int k0 = 0; k0 < ICic - 32; k0 += 32) {
      v16bf an = frag_a(WT + k0 + 32, ICic, lane);
#pragma unroll
      for (int nt = 0; nt < 4; ++nt)
        acc[nt] = wmma_bf16(a, frag_b(pb + (nt * 16) * 136 + k0, 136, lane),
                            acc[nt]);
      a = an;
    }
#pragma unroll
    for (int nt = 0; nt < 4; ++nt)
      acc[nt] = wmma_bf16(
          a, frag_b(pb + (nt * 16) * 136 + (ICic - 32), 136, lane), acc[nt]);
    const float* bias = outq ? btq : bts;
    bf16* T = outq ? T_q : T_s;
    float s1[8], s2[8];
#pragma unroll
    for (int r = 0; r < 8; ++r) { s1[r] = 0.f; s2[r] = 0.f; }
#pragma unroll
    for (int nt = 0; nt < 4; ++nt) {
      int n = n0 + nt * 16 + l15;
#pragma unroll
      for (int r = 0; r < 8; ++r) {
        int c = mt * 16 + r + 8 * g;
        float v = acc[nt][r] + bias[c];
        T[((size_t)b * Cc + c) * Nn + n] = (bf16)v;
        s1[r] += v;
        s2[r] += v * v;
      }
    }
#pragma unroll
    for (int r = 0; r < 8; ++r) {
#pragma unroll
      for (int m = 1; m < 16; m <<= 1) {
        s1[r] += __shfl_xor(s1[r], m, 32);
        s2[r] += __shfl_xor(s2[r], m, 32);
      }
      if (l15 == 0) {  // lanes 0 and 16: distinct channels -> single writer
        int c = mt * 16 + r + 8 * g;
        st[outq * 2 * Cc + c] = s1[r];
        st[outq * 2 * Cc + Cc + c] = s2[r];
      }
    }
  }
  __syncthreads();
  int wg = b * (Nn / 64) + blockIdx.x;
  for (int i = tid; i < 4 * Cc; i += 256)
    stats_part[(size_t)i * NWG_K3 + wg] = st[i];
}

// ---------------------------------------------------------------------------
// K4: deterministic BN stat reduction + finalize -> scale/shift per channel
// ---------------------------------------------------------------------------
__global__ void bn_finalize_kernel(const float* __restrict__ stats_part,
                                   const float* __restrict__ gts,
                                   const float* __restrict__ bets,
                                   const float* __restrict__ gtq,
                                   const float* __restrict__ betq,
                                   float* __restrict__ coef) {
  int c = threadIdx.x;
  if (c >= Cc) return;
  float acc[4];
#pragma unroll
  for (int k = 0; k < 4; ++k) {
    const float* p = stats_part + (size_t)(k * Cc + c) * NWG_K3;
    float s = 0.f;
    for (int w = 0; w < NWG_K3; ++w) s += p[w];
    acc[k] = s;
  }
  const float inv = 1.f / (float)(Bb * Nn);
  float mu = acc[0] * inv;
  float var = acc[1] * inv - mu * mu;
  float sc = gts[c] * rsqrtf(var + 1e-5f);
  coef[c] = sc;
  coef[Cc + c] = bets[c] - mu * sc;
  mu = acc[2] * inv;
  var = acc[3] * inv - mu * mu;
  sc = gtq[c] * rsqrtf(var + 1e-5f);
  coef[2 * Cc + c] = sc;
  coef[3 * Cc + c] = betq[c] - mu * sc;
}

// ---------------------------------------------------------------------------
// K5: token-mean pooling of q (y=0) and s (y=1). Grid (B*C, 2).
// ---------------------------------------------------------------------------
__global__ __launch_bounds__(256) void pool_kernel(const float* __restrict__ q,
                                                   const float* __restrict__ s,
                                                   float* __restrict__ pooled) {
  int id = blockIdx.x;
  const float* X = blockIdx.y ? s : q;
  float* P = pooled + (size_t)blockIdx.y * Bb * Cc;
  const float* row = X + (size_t)id * Nn;
  float sum = 0.f;
  for (int i = threadIdx.x; i < Nn; i += 256) sum += row[i];
#pragma unroll
  for (int m = 1; m < 32; m <<= 1) sum += __shfl_xor(sum, m, 32);
  __shared__ float wsum[8];
  if ((threadIdx.x & 31) == 0) wsum[threadIdx.x >> 5] = sum;
  __syncthreads();
  if (threadIdx.x == 0) {
    float t = 0.f;
    for (int i = 0; i < 8; ++i) t += wsum[i];
    P[id] = t * (1.f / (float)Nn);
  }
}

// ---------------------------------------------------------------------------
// K6: channel gate MLP (320->80 relu ->320 sigmoid). Grid (B, 2).
// ---------------------------------------------------------------------------
__global__ __launch_bounds__(256) void gate_kernel(
    const float* __restrict__ pooled, const float* __restrict__ Wg1,
    const float* __restrict__ bg1, const float* __restrict__ Wg2,
    const float* __restrict__ bg2, float* __restrict__ gates) {
  __shared__ float pool_l[Cc];
  __shared__ float h[80];
  int b = blockIdx.x;
  const float* P = pooled + (size_t)blockIdx.y * Bb * Cc + b * Cc;
  float* G = gates + (size_t)blockIdx.y * Bb * Cc + b * Cc;
  for (int i = threadIdx.x; i < Cc; i += 256) pool_l[i] = P[i];
  __syncthreads();
  if (threadIdx.x < 80) {
    float a = bg1[threadIdx.x];
    for (int c = 0; c < Cc; ++c) a += pool_l[c] * Wg1[c * 80 + threadIdx.x];
    h[threadIdx.x] = fmaxf(a, 0.f);
  }
  __syncthreads();
  for (int c = threadIdx.x; c < Cc; c += 256) {
    float a = bg2[c];
    for (int i = 0; i < 80; ++i) a += h[i] * Wg2[i * Cc + c];
    G[c] = 1.f / (1.f + __expf(-a));
  }
}

// ---------------------------------------------------------------------------
// K7: E = gate * (scale*T + shift) + residual; writes E_q then E_s (fp32)
// ---------------------------------------------------------------------------
__global__ __launch_bounds__(256) void final_kernel(
    const float* __restrict__ q, const float* __restrict__ s,
    const bf16* __restrict__ T_q, const bf16* __restrict__ T_s,
    const float* __restrict__ coef, const float* __restrict__ gates,
    float* __restrict__ out) {
  const size_t TOT = (size_t)Bb * Cc * Nn;
  size_t idx = ((size_t)blockIdx.x * 256 + threadIdx.x) * 4;
  if (idx >= TOT) return;
  size_t c = (idx >> 12) % Cc;
  size_t b = idx / ((size_t)Cc * Nn);
  float scs = coef[c], shs = coef[Cc + c];
  float scq = coef[2 * Cc + c], shq = coef[3 * Cc + c];
  float gq = gates[b * Cc + c];
  float gs = gates[(size_t)Bb * Cc + b * Cc + c];
  float4 q4 = *(const float4*)(q + idx);
  float4 s4 = *(const float4*)(s + idx);
  float qv[4] = {q4.x, q4.y, q4.z, q4.w};
  float sv[4] = {s4.x, s4.y, s4.z, s4.w};
  float oq[4], os[4];
#pragma unroll
  for (int i = 0; i < 4; ++i) {
    float tq = (float)T_q[idx + i];
    float ts = (float)T_s[idx + i];
    oq[i] = gq * (scq * tq + shq) + qv[i];
    os[i] = gs * (scs * ts + shs) + sv[i];
  }
  *(float4*)(out + idx) = make_float4(oq[0], oq[1], oq[2], oq[3]);
  *(float4*)(out + TOT + idx) = make_float4(os[0], os[1], os[2], os[3]);
}

// ---------------------------------------------------------------------------
extern "C" void kernel_launch(void* const* d_in, const int* in_sizes, int n_in,
                              void* d_out, int out_size, void* d_ws, size_t ws_size,
                              hipStream_t stream) {
  (void)in_sizes; (void)n_in; (void)out_size; (void)ws_size;
  const float* q   = (const float*)d_in[0];
  const float* s   = (const float*)d_in[1];
  const float* Wv  = (const float*)d_in[2];
  const float* bv  = (const float*)d_in[3];
  const float* Wk  = (const float*)d_in[4];
  const float* bk  = (const float*)d_in[5];
  const float* Wqp = (const float*)d_in[6];
  const float* bqp = (const float*)d_in[7];
  const float* Wts = (const float*)d_in[8];
  const float* bts = (const float*)d_in[9];
  const float* gts = (const float*)d_in[10];
  const float* bets = (const float*)d_in[11];
  const float* Wtq = (const float*)d_in[12];
  const float* btq = (const float*)d_in[13];
  const float* gtq = (const float*)d_in[14];
  const float* betq = (const float*)d_in[15];
  const float* Wg1 = (const float*)d_in[16];
  const float* bg1 = (const float*)d_in[17];
  const float* Wg2 = (const float*)d_in[18];
  const float* bg2 = (const float*)d_in[19];
  float* out = (float*)d_out;

  char* w = (char*)d_ws;
  size_t off = 0;
  auto take = [&](size_t bytes) -> char* {
    char* p = w + off;
    off += (bytes + 255) & ~(size_t)255;
    return p;
  };
  const size_t BIN = (size_t)Bb * ICic * Nn;
  const size_t BCN = (size_t)Bb * Cc * Nn;
  bf16* WvT  = (bf16*)take(ICic * Cc * 2);
  bf16* WkT  = (bf16*)take(ICic * Cc * 2);
  bf16* WqpT = (bf16*)take(ICic * Cc * 2);
  bf16* WtsT = (bf16*)take(Cc * ICic * 2);
  bf16* WtqT = (bf16*)take(Cc * ICic * 2);
  bf16* vqT  = (bf16*)take(BIN * 2);
  bf16* vsT  = (bf16*)take(BIN * 2);
  bf16* kx   = (bf16*)take(BIN * 2);
  bf16* qx   = (bf16*)take(BIN * 2);
  float* Apart = (float*)take((size_t)Bb * KCHUNK * ICic * ICic * 4);
  float* Aws = (float*)take((size_t)Bb * ICic * ICic * 4);
  bf16* atts = (bf16*)take((size_t)Bb * ICic * ICic * 2);
  bf16* attq = (bf16*)take((size_t)Bb * ICic * ICic * 2);
  bf16* Ts   = (bf16*)take(BCN * 2);
  bf16* Tq   = (bf16*)take(BCN * 2);
  float* stats_part = (float*)take((size_t)4 * Cc * NWG_K3 * 4);
  float* coef   = (float*)take(4 * Cc * 4);
  float* pooled = (float*)take(2 * Bb * Cc * 4);
  float* gates  = (float*)take(2 * Bb * Cc * 4);

  // K0: weight transpose + bf16 convert
  transpose_bf16_kernel<<<160, 256, 0, stream>>>(Wv, WvT, Cc, ICic);
  transpose_bf16_kernel<<<160, 256, 0, stream>>>(Wk, WkT, Cc, ICic);
  transpose_bf16_kernel<<<160, 256, 0, stream>>>(Wqp, WqpT, Cc, ICic);
  transpose_bf16_kernel<<<160, 256, 0, stream>>>(Wts, WtsT, ICic, Cc);
  transpose_bf16_kernel<<<160, 256, 0, stream>>>(Wtq, WtqT, ICic, Cc);

  // K1: projections (v_q,q_x from q) and (v_s,k_x from s)
  dim3 gp(Nn / 64, Bb);
  proj_kernel<<<gp, 256, 0, stream>>>(q, WvT, bv, vqT, WqpT, bqp, qx);
  proj_kernel<<<gp, 256, 0, stream>>>(s, WvT, bv, vsT, WkT, bk, kx);

  // K2: split-K attention logits + reduce + dual softmax
  attn_logits_kernel<<<dim3(Bb, KCHUNK), 256, 0, stream>>>(kx, qx, Apart);
  reduce_A_kernel<<<(Bb * ICic * ICic + 255) / 256, 256, 0, stream>>>(Apart, Aws);
  softmax_kernel<<<dim3(Bb, 2), 128, 0, stream>>>(Aws, atts, attq);

  // K3: fused attention-apply + Trans linear + BN partials
  apply_trans_kernel<<<gp, 256, 0, stream>>>(atts, attq, vsT, vqT, WtsT, bts,
                                             WtqT, btq, Ts, Tq, stats_part);

  // K4: BN deterministic reduce + finalize
  bn_finalize_kernel<<<1, Cc, 0, stream>>>(stats_part, gts, bets, gtq, betq, coef);

  // K5/K6: channel gate
  pool_kernel<<<dim3(Bb * Cc, 2), 256, 0, stream>>>(q, s, pooled);
  gate_kernel<<<dim3(Bb, 2), 256, 0, stream>>>(pooled, Wg1, bg1, Wg2, bg2, gates);

  // K7: BN affine + gate + residual -> E_q, E_s
  size_t quads = BCN / 4;
  final_kernel<<<(unsigned)((quads + 255) / 256), 256, 0, stream>>>(
      q, s, Tq, Ts, coef, gates, out);
}
